// GraphTransformerLayer_30906584662329
// MI455X (gfx1250) — compile-verified, matrix operand order checked by reading, not run on hardware
//
#include <hip/hip_runtime.h>

#define NNODES 50000
#define DIM 128
#define NHEADS 8
#define HD 16
#define ROWTILES (NNODES / 16)   // 3125 exactly

typedef __attribute__((ext_vector_type(16))) _Float16 v16h;
typedef __attribute__((ext_vector_type(8)))  float    v8f;

// CDNA5 16-bit A-matrix 16x32 layout (ISA 7.12.2):
// lanes 0-15  (sel=0): halves 0..7 -> K=0..7,  halves 8..15 -> K=16..23
// lanes 16-31 (sel=1): halves 0..7 -> K=8..15, halves 8..15 -> K=24..31
__device__ __forceinline__ int kmap(int i, int sel) {
    int base = sel ? 8 : 0;
    return (i < 8) ? (base + i) : (16 + base + (i - 8));
}

// Load one 16x32 A fragment from f32 row-major source (vectorized float4)
__device__ __forceinline__ v16h load_a_f32(const float* __restrict__ src, int lda) {
    int lane = threadIdx.x & 31;
    int m = lane & 15, sel = lane >> 4;
    const float4* p = (const float4*)(src + m * lda);
    float4 u0 = p[sel * 2 + 0];      // K = sel*8 .. sel*8+3
    float4 u1 = p[sel * 2 + 1];      // K = sel*8+4 .. sel*8+7
    float4 u2 = p[4 + sel * 2 + 0];  // K = 16+sel*8 ..
    float4 u3 = p[4 + sel * 2 + 1];
    v16h a;
    a[0]  = (_Float16)u0.x; a[1]  = (_Float16)u0.y; a[2]  = (_Float16)u0.z; a[3]  = (_Float16)u0.w;
    a[4]  = (_Float16)u1.x; a[5]  = (_Float16)u1.y; a[6]  = (_Float16)u1.z; a[7]  = (_Float16)u1.w;
    a[8]  = (_Float16)u2.x; a[9]  = (_Float16)u2.y; a[10] = (_Float16)u2.z; a[11] = (_Float16)u2.w;
    a[12] = (_Float16)u3.x; a[13] = (_Float16)u3.y; a[14] = (_Float16)u3.z; a[15] = (_Float16)u3.w;
    return a;
}

// Load one 16x32 A fragment from an f16 LDS tile [16][DIM]
// (per-lane halves are two contiguous 16B runs -> merges into ds_load_b128 x2)
__device__ __forceinline__ v16h load_a_lds(const _Float16* tile, int k0) {
    int lane = threadIdx.x & 31;
    int m = lane & 15, sel = lane >> 4;
    const _Float16* row = tile + m * DIM + k0;
    v16h a;
#pragma unroll
    for (int i = 0; i < 16; ++i) a[i] = row[kmap(i, sel)];
    return a;
}

// Pre-packed B fragments: per fragment (j,kk), lane l owns 16 contiguous halves
// (32 bytes) -> a single vector load (2 x global_load_b128) at GEMM time.
__device__ __forceinline__ v16h load_b_packed(const _Float16* __restrict__ P, int j, int kk) {
    int lane = threadIdx.x & 31;
    return *(const v16h*)(P + ((size_t)(j * 4 + kk) * 32 + lane) * 16);
}

// ---------------------------------------------------------------------------
// Kernel 0: pack one f32 weight matrix [128][128] into WMMA B-fragment f16
// layout. One wave per (j,kk) fragment; 32 fragments per matrix.
// ---------------------------------------------------------------------------
__global__ __launch_bounds__(32) void pack_w(const float* __restrict__ W,
                                             _Float16* __restrict__ P) {
    int frag = blockIdx.x;          // j*4 + kk
    int j = frag >> 2, kk = frag & 3;
    int lane = threadIdx.x & 31;
    int n = lane & 15, sel = lane >> 4;
    _Float16* dstp = P + ((size_t)frag * 32 + lane) * 16;
#pragma unroll
    for (int i = 0; i < 16; ++i)
        dstp[i] = (_Float16)W[(size_t)(kk * 32 + kmap(i, sel)) * DIM + j * 16 + n];
}

// ---------------------------------------------------------------------------
// Kernel 1: fused Q/K/V projection. One wave (32 threads) per 16-row tile.
// B fragments double-buffered across the column-tile loop so global loads
// overlap the WMMAs instead of serializing load->wait->wmma.
// ---------------------------------------------------------------------------
__global__ __launch_bounds__(32) void qkv_gemm(
        const float* __restrict__ x,
        const _Float16* __restrict__ Pq, const float* __restrict__ bq,
        const _Float16* __restrict__ Pk, const float* __restrict__ bk,
        const _Float16* __restrict__ Pv, const float* __restrict__ bv,
        float* __restrict__ Q, float* __restrict__ K, float* __restrict__ V) {
    int row0 = blockIdx.x * 16;
    int lane = threadIdx.x & 31;

    v16h a[4];
#pragma unroll
    for (int kk = 0; kk < 4; ++kk)
        a[kk] = load_a_f32(x + (size_t)row0 * DIM + kk * 32, DIM);

    const _Float16* Ps[3] = {Pq, Pk, Pv};
    const float*    bs[3] = {bq, bk, bv};
    float*          Os[3] = {Q, K, V};

    int n  = lane & 15;
    int mb = (lane >> 4) * 8;

#pragma unroll
    for (int mtx = 0; mtx < 3; ++mtx) {
        const _Float16* P    = Ps[mtx];
        const float*    bias = bs[mtx];
        float*          out  = Os[mtx];

        v16h b[4];
#pragma unroll
        for (int kk = 0; kk < 4; ++kk) b[kk] = load_b_packed(P, 0, kk);

#pragma unroll
        for (int j = 0; j < 8; ++j) {
            v16h bn[4];
            if (j < 7) {
#pragma unroll
                for (int kk = 0; kk < 4; ++kk) bn[kk] = load_b_packed(P, j + 1, kk);
            }
            v8f acc = {};
#pragma unroll
            for (int kk = 0; kk < 4; ++kk)
                acc = __builtin_amdgcn_wmma_f32_16x16x32_f16(
                        false, a[kk], false, b[kk], (short)0, acc, false, false);

            int col = j * 16 + n;
            float bb = bias[col];
#pragma unroll
            for (int i = 0; i < 8; ++i)
                out[(size_t)(row0 + mb + i) * DIM + col] = acc[i] + bb;

#pragma unroll
            for (int kk = 0; kk < 4; ++kk) b[kk] = bn[kk];
        }
    }
}

// ---------------------------------------------------------------------------
// Kernel 2: zero the wV accumulator
// ---------------------------------------------------------------------------
__global__ void zero_f32(float* __restrict__ p, int n) {
    int i = blockIdx.x * blockDim.x + threadIdx.x;
    if (i < n) p[i] = 0.0f;
}

// ---------------------------------------------------------------------------
// Kernel 3: per-(edge, head) attention score + scatter-add into wV
// ---------------------------------------------------------------------------
__global__ void edge_attn(const float* __restrict__ Q, const float* __restrict__ K,
                          const float* __restrict__ V, const int* __restrict__ src,
                          const int* __restrict__ dst, float* __restrict__ wV, int E) {
    int t = blockIdx.x * blockDim.x + threadIdx.x;
    if (t >= E * NHEADS) return;
    int e = t >> 3;
    int h = t & 7;
    int s = src[e];
    int d = dst[e];

    const float4* kp = (const float4*)(K + (size_t)s * DIM + h * HD);
    const float4* qp = (const float4*)(Q + (size_t)d * DIM + h * HD);
    float dot = 0.0f;
#pragma unroll
    for (int i = 0; i < 4; ++i) {
        float4 kv = kp[i], qv = qp[i];
        dot += kv.x * qv.x + kv.y * qv.y + kv.z * qv.z + kv.w * qv.w;
    }
    dot *= 0.25f;                           // 1/sqrt(16)
    dot = fminf(5.0f, fmaxf(-5.0f, dot));   // clamp(-5, 5)
    float sc = __expf(dot);

    const float4* vp = (const float4*)(V + (size_t)s * DIM + h * HD);
    float* o = wV + (size_t)d * DIM + h * HD;
#pragma unroll
    for (int i = 0; i < 4; ++i) {
        float4 vv = vp[i];
        unsafeAtomicAdd(o + i * 4 + 0, vv.x * sc);
        unsafeAtomicAdd(o + i * 4 + 1, vv.y * sc);
        unsafeAtomicAdd(o + i * 4 + 2, vv.z * sc);
        unsafeAtomicAdd(o + i * 4 + 3, vv.w * sc);
    }
}

// ---------------------------------------------------------------------------
// Kernel 4: residual + LN1 -> LN2 -> Wo GEMM (WMMA from LDS) -> ReLU + residual
// One 16-row tile per block of 128 threads (4 waves). Each wave's two Wo
// B-column-tiles are preloaded BEFORE the LN phase so the global loads are in
// flight under ~100 VALU ops of LayerNorm work.
// ---------------------------------------------------------------------------
__global__ __launch_bounds__(128) void fused_ln_out(
        const float* __restrict__ x, const float* __restrict__ wV,
        const _Float16* __restrict__ Po, const float* __restrict__ bo,
        const float* __restrict__ g1, const float* __restrict__ b1,
        const float* __restrict__ g2, const float* __restrict__ b2,
        float* __restrict__ out) {
    __shared__ float    h2s[16 * DIM];   // post-LN1 tile (residual for epilogue)
    __shared__ _Float16 gs[16 * DIM];    // post-LN2 tile (A matrix for WMMA)

    int row0 = blockIdx.x * 16;
    int t = threadIdx.x;
    int r  = t >> 3;          // row in tile (8 threads per row; contiguous in wave)
    int c0 = (t & 7) * 16;    // 16 columns per thread
    int wave = t >> 5;
    int lane = t & 31;

    // --- preload this wave's B fragments (independent of LDS/LN work) ---
    v16h b0[4], b1f[4];
#pragma unroll
    for (int kk = 0; kk < 4; ++kk) {
        b0[kk]  = load_b_packed(Po, wave * 2 + 0, kk);
        b1f[kk] = load_b_packed(Po, wave * 2 + 1, kk);
    }

    // --- residual + LN1 ---
    float v[16];
    float s = 0.0f, ss = 0.0f;
#pragma unroll
    for (int i = 0; i < 16; ++i) {
        size_t idx = (size_t)(row0 + r) * DIM + c0 + i;
        float val = x[idx] + wV[idx];
        v[i] = val; s += val; ss += val * val;
    }
#pragma unroll
    for (int m = 1; m < 8; m <<= 1) {     // reduce within the 8 lanes of a row
        s  += __shfl_xor(s,  m, 32);
        ss += __shfl_xor(ss, m, 32);
    }
    float mean = s * (1.0f / DIM);
    float var  = ss * (1.0f / DIM) - mean * mean;
    float rstd = rsqrtf(var + 1e-5f);

    // --- LN2 on LN1 output ---
    float s2 = 0.0f, ss2 = 0.0f;
#pragma unroll
    for (int i = 0; i < 16; ++i) {
        int c = c0 + i;
        float h = (v[i] - mean) * rstd * g1[c] + b1[c];
        v[i] = h;
        h2s[r * DIM + c] = h;
        s2 += h; ss2 += h * h;
    }
#pragma unroll
    for (int m = 1; m < 8; m <<= 1) {
        s2  += __shfl_xor(s2,  m, 32);
        ss2 += __shfl_xor(ss2, m, 32);
    }
    float mean2 = s2 * (1.0f / DIM);
    float var2  = ss2 * (1.0f / DIM) - mean2 * mean2;
    float rstd2 = rsqrtf(var2 + 1e-5f);
#pragma unroll
    for (int i = 0; i < 16; ++i) {
        int c = c0 + i;
        gs[r * DIM + c] = (_Float16)((v[i] - mean2) * rstd2 * g2[c] + b2[c]);
    }
    __syncthreads();

    // --- Wo GEMM + bias + ReLU + residual epilogue ---
    v16h a[4];
#pragma unroll
    for (int kk = 0; kk < 4; ++kk) a[kk] = load_a_lds(gs, kk * 32);

    int n  = lane & 15;
    int mb = (lane >> 4) * 8;

#pragma unroll
    for (int jj = 0; jj < 2; ++jj) {      // 8 column tiles / 4 waves
        int j = wave * 2 + jj;
        v8f acc = {};
#pragma unroll
        for (int kk = 0; kk < 4; ++kk) {
            v16h b = (jj == 0) ? b0[kk] : b1f[kk];
            acc = __builtin_amdgcn_wmma_f32_16x16x32_f16(
                    false, a[kk], false, b, (short)0, acc, false, false);
        }
        int col = j * 16 + n;
        float bb = bo[col];
#pragma unroll
        for (int i = 0; i < 8; ++i) {
            float relu = fmaxf(acc[i] + bb, 0.0f);
            out[(size_t)(row0 + mb + i) * DIM + col] = h2s[(mb + i) * DIM + col] + relu;
        }
    }
}

// ---------------------------------------------------------------------------
extern "C" void kernel_launch(void* const* d_in, const int* in_sizes, int n_in,
                              void* d_out, int out_size, void* d_ws, size_t ws_size,
                              hipStream_t stream) {
    const float* x  = (const float*)d_in[0];
    const int*   src = (const int*)d_in[1];
    const int*   dst = (const int*)d_in[2];
    const float* Wq = (const float*)d_in[3];
    const float* bq = (const float*)d_in[4];
    const float* Wk = (const float*)d_in[5];
    const float* bk = (const float*)d_in[6];
    const float* Wv = (const float*)d_in[7];
    const float* bv = (const float*)d_in[8];
    const float* Wo = (const float*)d_in[9];
    const float* bo = (const float*)d_in[10];
    const float* g1 = (const float*)d_in[11];
    const float* b1 = (const float*)d_in[12];
    const float* g2 = (const float*)d_in[13];
    const float* b2 = (const float*)d_in[14];
    float* out = (float*)d_out;

    int E = in_sizes[1];
    size_t nd = (size_t)NNODES * DIM;
    float* Q  = (float*)d_ws;
    float* K  = Q + nd;
    float* V  = K + nd;
    float* wV = V + nd;
    // packed f16 weight fragments: 32 frags * 32 lanes * 16 halves = 16384 halves each
    _Float16* Pq = (_Float16*)(wV + nd);
    _Float16* Pk = Pq + 32 * 32 * 16;
    _Float16* Pv = Pk + 32 * 32 * 16;
    _Float16* Po = Pv + 32 * 32 * 16;

    pack_w<<<32, 32, 0, stream>>>(Wq, Pq);
    pack_w<<<32, 32, 0, stream>>>(Wk, Pk);
    pack_w<<<32, 32, 0, stream>>>(Wv, Pv);
    pack_w<<<32, 32, 0, stream>>>(Wo, Po);

    zero_f32<<<(int)((nd + 255) / 256), 256, 0, stream>>>(wV, (int)nd);
    qkv_gemm<<<ROWTILES, 32, 0, stream>>>(x, Pq, bq, Pk, bk, Pv, bv, Q, K, V);
    int tot = E * NHEADS;
    edge_attn<<<(tot + 255) / 256, 256, 0, stream>>>(Q, K, V, src, dst, wV, E);
    fused_ln_out<<<ROWTILES, 128, 0, stream>>>(x, wV, Po, bo, g1, b1, g2, b2, out);
}
